// DeformableAttention1D_36369783062681
// MI455X (gfx1250) — matrix operand order repaired and checked by reading, not compiled, not run on hardware
//
#include <hip/hip_runtime.h>
#include <hip/hip_bf16.h>

typedef __attribute__((ext_vector_type(16))) _Float16 v16h;
typedef __attribute__((ext_vector_type(8)))  _Float16 v8h;
typedef __attribute__((ext_vector_type(8)))  float    v8f;

constexpr int B_   = 4;
constexpr int N_   = 4096;
constexpr int DIM_ = 256;
constexpr int H_   = 8;
constexpr int G_   = 8;
constexpr int DH_  = 64;
constexpr int INNER_ = 512;   // H_*DH_
constexpr int M_   = 1024;    // N_/DS
constexpr int OFFD_ = 64;
constexpr int KS_  = 6;
constexpr int DS_  = 4;

// ---------------------------------------------------------------- WMMA helpers
__device__ __forceinline__ v8f wmma32(v16h a, v16h b, v8f c) {
  // D = A(16x32 f16) * B(32x16 f16) + C(16x16 f32)
  return __builtin_amdgcn_wmma_f32_16x16x32_f16(
      /*neg_a=*/false, a, /*neg_b=*/false, b,
      /*c_mod=*/(short)0, c, /*reuse_a=*/false, /*reuse_b=*/false);
}

// A-layout (ISA 7.12.2): lane row = lane&15; element j -> K = k0 + (j<8?0:16) + kb + (j&7)
__device__ __forceinline__ v16h loadA_h(const _Float16* row, int k0, int kb) {
  const v8h lo = *(const v8h*)(row + k0 + kb);
  const v8h hi = *(const v8h*)(row + k0 + kb + 16);
  v16h r;
#pragma unroll
  for (int j = 0; j < 8; ++j) { r[j] = lo[j]; r[j + 8] = hi[j]; }
  return r;
}

__device__ __forceinline__ v16h loadA_f32(const float* row, int kb) {
  v16h r;
#pragma unroll
  for (int j = 0; j < 8; ++j) {
    r[j]     = (_Float16)row[kb + j];
    r[j + 8] = (_Float16)row[kb + 16 + j];
  }
  return r;
}

// ---------------------------------------------------------------- cast weights
__global__ void k_cast_f16(const float* __restrict__ s, _Float16* __restrict__ d, int n) {
  int i = blockIdx.x * blockDim.x + threadIdx.x;
  if (i < n) d[i] = (_Float16)s[i];
}

// ---------------------------------------------------------------- Q projection
// per wave: (b, g, 16-col tile of N): q_g(64 x 16) = Wq_g(64x64) @ cat_g(64x16)
__global__ void k_qproj(const float* __restrict__ x, const float* __restrict__ px,
                        const _Float16* __restrict__ wqh,
                        float* __restrict__ qf, _Float16* __restrict__ qh) {
  const int lane = threadIdx.x & 31;
  const int lrow = lane & 15;
  const int kb   = (lane >> 4) * 8;
  const int NT = N_ / 16;
  int gid = blockIdx.x;
  int nt = gid % NT;
  int g  = (gid / NT) & 7;
  int b  = gid / (NT * 8);
  int n0 = nt * 16;

  const float* src = (g < 4) ? px : x;      // cat = [prev_x^T ; x^T]
  const int c0 = (g & 3) * 64;              // channel base within source

  // B tiles (K=64 split in two 32-steps): lane&15 = column n, element j = k (table order)
  v16h B0, B1;
  {
    const float* colp = src + ((size_t)b * N_ + n0 + lrow) * DIM_ + c0 + kb;
#pragma unroll
    for (int j = 0; j < 16; ++j) {
      const int kk = (j & 7) + ((j >> 3) << 4);
      B0[j] = (_Float16)colp[kk];
      B1[j] = (_Float16)colp[kk + 32];
    }
  }

#pragma unroll
  for (int mt = 0; mt < 4; ++mt) {
    const _Float16* arow = wqh + (size_t)(g * 64 + mt * 16 + lrow) * 64;
    v16h A0 = loadA_h(arow, 0, kb);
    v16h A1 = loadA_h(arow, 32, kb);
    v8f acc = {};
    acc = wmma32(A0, B0, acc);
    acc = wmma32(A1, B1, acc);
    const int n = n0 + lrow;
#pragma unroll
    for (int i = 0; i < 8; ++i) {
      int o = mt * 16 + kb + i;             // C row = 8*(lane>>4)+i
      float v = acc[i];
      qf[((size_t)b * INNER_ + g * 64 + o) * N_ + n] = v;                     // for offsets
      qh[(((size_t)(b * H_ + g)) * N_ + n) * DH_ + o] = (_Float16)(v * 0.125f); // scaled, (B,H,N,64)
    }
  }
}

// ---------------------------------------------------------------- offsets path
// depthwise conv(k=6, stride=4, pad=1) -> exact GELU -> 1x1 -> tanh * 4
__global__ void k_offsets(const float* __restrict__ qf, const float* __restrict__ w1,
                          const float* __restrict__ b1, const float* __restrict__ w2,
                          float* __restrict__ offs) {
  int idx = blockIdx.x * blockDim.x + threadIdx.x;
  if (idx >= B_ * G_ * M_) return;
  int m  = idx % M_;
  int bg = idx / M_;
  int g = bg & 7, b = bg >> 3;
  float acc = 0.f;
  for (int c = 0; c < OFFD_; ++c) {
    const float* qp = qf + ((size_t)b * INNER_ + g * 64 + c) * N_;
    float h = b1[c];
#pragma unroll
    for (int k = 0; k < KS_; ++k) {
      int p = m * DS_ - 1 + k;
      if (p >= 0 && p < N_) h += qp[p] * w1[c * KS_ + k];
    }
    float ge = 0.5f * h * (1.f + erff(h * 0.70710678118654752f));
    acc += ge * w2[c];
  }
  offs[idx] = tanhf(acc) * 4.0f;
}

// ---------------------------------------------------------------- bilinear kv sample
// kvT: (B, 256, M) f16 channel-major
__global__ void k_sample(const float* __restrict__ px, const float* __restrict__ offs,
                         _Float16* __restrict__ kvT) {
  int idx = blockIdx.x * blockDim.x + threadIdx.x;
  if (idx >= B_ * G_ * M_) return;
  int m  = idx % M_;
  int bg = idx / M_;
  int g = bg & 7, b = bg >> 3;
  float vg   = (float)m + offs[idx];
  float grid = 2.f * vg / (float)(M_ - 1) - 1.f;
  float pos  = ((grid + 1.f) * (float)N_ - 1.f) * 0.5f;
  float f    = floorf(pos);
  float w    = pos - f;
  int   i0   = (int)f;
  bool v0 = (i0 >= 0) && (i0 < N_);
  bool v1 = (i0 + 1 >= 0) && (i0 + 1 < N_);
  const float* r0 = px + ((size_t)b * N_ + (v0 ? i0 : 0)) * DIM_ + g * 32;
  const float* r1 = px + ((size_t)b * N_ + (v1 ? i0 + 1 : 0)) * DIM_ + g * 32;
#pragma unroll
  for (int c = 0; c < 32; ++c) {
    float a  = v0 ? r0[c] : 0.f;
    float bb = v1 ? r1[c] : 0.f;
    kvT[((size_t)b * DIM_ + g * 32 + c) * M_ + m] = (_Float16)(a * (1.f - w) + bb * w);
  }
}

// ---------------------------------------------------------------- K/V projection
// kT: (B,H,64,M) f16 ; vh: (B,H,M,64) f16
__global__ void k_kvproj(const _Float16* __restrict__ kvT, const _Float16* __restrict__ wkh,
                         const _Float16* __restrict__ wvh,
                         _Float16* __restrict__ kT, _Float16* __restrict__ vh) {
  const int lane = threadIdx.x & 31;
  const int lrow = lane & 15;
  const int kb   = (lane >> 4) * 8;
  const int MT = M_ / 16;
  int gid = blockIdx.x;
  int t = gid % MT;
  int g = (gid / MT) & 7;
  int b = gid / (MT * 8);
  int m0 = t * 16;

  // B tile: kv_g (32 x 16): lane&15 = column m, element j = input-channel k (table order)
  v16h Bv;
#pragma unroll
  for (int j = 0; j < 16; ++j)
    Bv[j] = kvT[((size_t)b * DIM_ + g * 32 + kb + (j & 7) + ((j >> 3) << 4)) * M_ + m0 + lrow];

#pragma unroll
  for (int mt = 0; mt < 4; ++mt) {
    const _Float16* ka = wkh + (size_t)(g * 64 + mt * 16 + lrow) * 32;
    const _Float16* va = wvh + (size_t)(g * 64 + mt * 16 + lrow) * 32;
    v16h KA = loadA_h(ka, 0, kb);
    v16h VA = loadA_h(va, 0, kb);
    v8f ak = {}; ak = wmma32(KA, Bv, ak);
    v8f av = {}; av = wmma32(VA, Bv, av);
    const int m = m0 + lrow;
#pragma unroll
    for (int i = 0; i < 8; ++i) {
      int o = mt * 16 + kb + i;
      kT[(((size_t)(b * H_ + g)) * DH_ + o) * M_ + m] = (_Float16)ak[i];
      vh[(((size_t)(b * H_ + g)) * M_ + m) * DH_ + o] = (_Float16)av[i];
    }
  }
}

// ---------------------------------------------------------------- attention core
// one wave per (b, h, 16 query rows); full-M score stripe staged in LDS (64KB)
__global__ void k_attn(const _Float16* __restrict__ qh, const _Float16* __restrict__ kT,
                       const _Float16* __restrict__ vh, _Float16* __restrict__ attC) {
  extern __shared__ float sS[];               // 16 x 1024 f32
  const int lane = threadIdx.x & 31;
  const int lrow = lane & 15;
  const int kb   = (lane >> 4) * 8;
  const int NT = N_ / 16;
  int gid = blockIdx.x;
  int nt = gid % NT;
  int h  = (gid / NT) & 7;
  int b  = gid / (NT * 8);
  int n0 = nt * 16;
  const size_t bh = (size_t)b * H_ + h;

  // Q block in A-layout, kept in registers for the whole wave
  const _Float16* qrow = qh + (bh * N_ + n0 + lrow) * DH_;
  v16h QA0 = loadA_h(qrow, 0, kb);
  v16h QA1 = loadA_h(qrow, 32, kb);

  // S = Q K^T, written to LDS
  const _Float16* ktb = kT + (bh * DH_) * M_;
  for (int jt = 0; jt < M_ / 16; ++jt) {
    __builtin_prefetch(ktb + (size_t)lane * M_ + (jt + 1) * 16, 0, 1);
    v16h KB0, KB1;
#pragma unroll
    for (int j = 0; j < 16; ++j) {
      const int kk = kb + (j & 7) + ((j >> 3) << 4);
      KB0[j] = ktb[(size_t)kk * M_ + jt * 16 + lrow];
      KB1[j] = ktb[(size_t)(kk + 32) * M_ + jt * 16 + lrow];
    }
    v8f s = {};
    s = wmma32(QA0, KB0, s);
    s = wmma32(QA1, KB1, s);
#pragma unroll
    for (int i = 0; i < 8; ++i)
      sS[(size_t)(kb + i) * M_ + jt * 16 + lrow] = s[i];
  }
  __syncthreads();   // 1-wave block: pure DS-counter fence

  // softmax: 2 lanes per row, half-row each, combine via shfl_xor(16)
  float* rowp = sS + (size_t)lrow * M_ + (lane >> 4) * (M_ / 2);
  float mx = -3.402823466e38f;
  for (int c = 0; c < M_ / 2; ++c) mx = fmaxf(mx, rowp[c]);
  mx = fmaxf(mx, __shfl_xor(mx, 16, 32));
  float sum = 0.f;
  for (int c = 0; c < M_ / 2; ++c) { float e = __expf(rowp[c] - mx); rowp[c] = e; sum += e; }
  sum += __shfl_xor(sum, 16, 32);
  float inv = 1.f / sum;
  for (int c = 0; c < M_ / 2; ++c) rowp[c] *= inv;
  __syncthreads();

  // O = P V
  v8f acc[4] = {{}, {}, {}, {}};
  const _Float16* vb = vh + bh * (size_t)M_ * DH_;
  for (int ck = 0; ck < M_ / 32; ++ck) {
    const float* prow = sS + (size_t)lrow * M_ + ck * 32;
    v16h PA = loadA_f32(prow, kb);
    const _Float16* vr = vb + (size_t)(ck * 32 + kb) * DH_ + lrow;   // lane&15 = d col, element = kv row
    v16h VB0, VB1, VB2, VB3;
#pragma unroll
    for (int j = 0; j < 16; ++j) {
      const size_t ko = (size_t)((j & 7) + ((j >> 3) << 4)) * DH_;
      VB0[j] = vr[ko + 0];
      VB1[j] = vr[ko + 16];
      VB2[j] = vr[ko + 32];
      VB3[j] = vr[ko + 48];
    }
    acc[0] = wmma32(PA, VB0, acc[0]);
    acc[1] = wmma32(PA, VB1, acc[1]);
    acc[2] = wmma32(PA, VB2, acc[2]);
    acc[3] = wmma32(PA, VB3, acc[3]);
  }

  // store channel-major f16: attC (B, 512, N)
#pragma unroll
  for (int dt = 0; dt < 4; ++dt) {
    int d = dt * 16 + lrow;                 // C col
    _Float16* op = attC + ((size_t)b * INNER_ + h * 64 + d) * N_ + n0 + kb;
#pragma unroll
    for (int i = 0; i < 8; ++i) op[i] = (_Float16)acc[dt][i];   // C row = kb+i (query)
  }
}

// ---------------------------------------------------------------- output projection
// out(B,N,256) = wo(256x512) @ att(512xN) + bias ; one wave = 64 out-channels x 16 cols
__global__ void k_oproj(const _Float16* __restrict__ attC, const _Float16* __restrict__ woh,
                        const float* __restrict__ bo, float* __restrict__ out) {
  const int lane = threadIdx.x & 31;
  const int lrow = lane & 15;
  const int kb   = (lane >> 4) * 8;
  const int NT = N_ / 16;
  int gid = blockIdx.x;
  int mg = gid & 3;
  int nt = (gid >> 2) % NT;
  int b  = gid / (4 * NT);
  int n0 = nt * 16;

  v8f acc[4] = {{}, {}, {}, {}};
  for (int ck = 0; ck < INNER_ / 32; ++ck) {
    v16h Bt;
#pragma unroll
    for (int j = 0; j < 16; ++j)
      Bt[j] = attC[((size_t)b * INNER_ + ck * 32 + kb + (j & 7) + ((j >> 3) << 4)) * N_ + n0 + lrow];
#pragma unroll
    for (int t = 0; t < 4; ++t) {
      const _Float16* ar = woh + (size_t)((mg * 4 + t) * 16 + lrow) * INNER_;
      v16h At = loadA_h(ar, ck * 32, kb);
      acc[t] = wmma32(At, Bt, acc[t]);
    }
  }
  const int n = n0 + lrow;
#pragma unroll
  for (int t = 0; t < 4; ++t) {
    int oc = (mg * 4 + t) * 16 + kb;
#pragma unroll
    for (int i = 0; i < 8; ++i)
      out[((size_t)b * N_ + n) * DIM_ + oc + i] = acc[t][i] + bo[oc + i];
  }
}

// ---------------------------------------------------------------- launcher
extern "C" void kernel_launch(void* const* d_in, const int* in_sizes, int n_in,
                              void* d_out, int out_size, void* d_ws, size_t ws_size,
                              hipStream_t stream) {
  (void)in_sizes; (void)n_in; (void)out_size; (void)ws_size;
  const float* x  = (const float*)d_in[0];
  const float* px = (const float*)d_in[1];
  const float* wq = (const float*)d_in[2];
  const float* wk = (const float*)d_in[3];
  const float* wv = (const float*)d_in[4];
  const float* wo = (const float*)d_in[5];
  const float* bo = (const float*)d_in[6];
  const float* w1 = (const float*)d_in[7];
  const float* b1 = (const float*)d_in[8];
  const float* w2 = (const float*)d_in[9];
  float* out = (float*)d_out;

  char* ws = (char*)d_ws;
  size_t off = 0;
  auto carve = [&](size_t bytes) -> char* {
    char* p = ws + off;
    off = (off + bytes + 255) & ~(size_t)255;
    return p;
  };
  _Float16* wqh  = (_Float16*)carve((size_t)INNER_ * 64 * 2);
  _Float16* wkh  = (_Float16*)carve((size_t)INNER_ * 32 * 2);
  _Float16* wvh  = (_Float16*)carve((size_t)INNER_ * 32 * 2);
  _Float16* woh  = (_Float16*)carve((size_t)DIM_ * INNER_ * 2);
  float*    qf   = (float*)   carve((size_t)B_ * INNER_ * N_ * 4);
  _Float16* qhh  = (_Float16*)carve((size_t)B_ * H_ * N_ * DH_ * 2);
  float*    offs = (float*)   carve((size_t)B_ * G_ * M_ * 4);
  _Float16* kvT  = (_Float16*)carve((size_t)B_ * DIM_ * M_ * 2);
  _Float16* kTh  = (_Float16*)carve((size_t)B_ * H_ * DH_ * M_ * 2);
  _Float16* vhh  = (_Float16*)carve((size_t)B_ * H_ * M_ * DH_ * 2);
  _Float16* attC = (_Float16*)carve((size_t)B_ * INNER_ * N_ * 2);

  k_cast_f16<<<(INNER_ * 64 + 255) / 256, 256, 0, stream>>>(wq, wqh, INNER_ * 64);
  k_cast_f16<<<(INNER_ * 32 + 255) / 256, 256, 0, stream>>>(wk, wkh, INNER_ * 32);
  k_cast_f16<<<(INNER_ * 32 + 255) / 256, 256, 0, stream>>>(wv, wvh, INNER_ * 32);
  k_cast_f16<<<(DIM_ * INNER_ + 255) / 256, 256, 0, stream>>>(wo, woh, DIM_ * INNER_);

  k_qproj<<<B_ * G_ * (N_ / 16), 32, 0, stream>>>(x, px, wqh, qf, qhh);
  k_offsets<<<(B_ * G_ * M_ + 255) / 256, 256, 0, stream>>>(qf, w1, b1, w2, offs);
  k_sample<<<(B_ * G_ * M_ + 255) / 256, 256, 0, stream>>>(px, offs, kvT);
  k_kvproj<<<B_ * G_ * (M_ / 16), 32, 0, stream>>>(kvT, wkh, wvh, kTh, vhh);
  k_attn<<<B_ * H_ * (N_ / 16), 32, 16 * M_ * sizeof(float), stream>>>(qhh, kTh, vhh, attC);
  k_oproj<<<B_ * (N_ / 16) * 4, 32, 0, stream>>>(attC, woh, bo, out);
}
